// MultiheadAttention_27608049779418
// MI455X (gfx1250) — matrix operand order repaired and checked by reading, not compiled, not run on hardware
//
#include <hip/hip_runtime.h>
#include <hip/hip_bf16.h>
#include <cstdint>
#include <cstddef>

// MI455X / gfx1250 multi-head attention (B=2,S=4096,E=1024,H=16,D=64)
// bf16 WMMA with f32 accumulation. Workspace requirement: 72 MB.

#define DEVI __device__ __forceinline__

typedef __attribute__((ext_vector_type(16))) __bf16   v16bf;
typedef __attribute__((ext_vector_type(8)))  __bf16   v8bf;
typedef __attribute__((ext_vector_type(8)))  float    v8f;
typedef __attribute__((ext_vector_type(4)))  float    v4f;
typedef __attribute__((ext_vector_type(8)))  unsigned v8u;
typedef __attribute__((ext_vector_type(4)))  unsigned v4u;

constexpr int Bc = 2;
constexpr int Sc = 4096;
constexpr int Ec = 1024;
constexpr int Hc = 16;
constexpr int Dc = 64;
constexpr int Mrows = Bc * Sc;   // 8192 token rows

// ---------- bf16 helpers ----------

// Round two f32 to bf16 (RNE) and pack into one dword with a single
// v_perm_b32 byte-select (low half = a, high half = b).
DEVI unsigned pk2bf(float a, float b) {
  unsigned ua = __builtin_bit_cast(unsigned, a);
  unsigned ub = __builtin_bit_cast(unsigned, b);
  ua += 0x7FFFu + ((ua >> 16) & 1u);
  ub += 0x7FFFu + ((ub >> 16) & 1u);
  return __builtin_amdgcn_perm(ub, ua, 0x07060302u);  // {ub[3],ub[2],ua[3],ua[2]}
}

DEVI __bf16 f2bf(float f) {
  unsigned u = __builtin_bit_cast(unsigned, f);
  unsigned short hs = (unsigned short)((u + 0x7FFFu + ((u >> 16) & 1u)) >> 16);
  return __builtin_bit_cast(__bf16, hs);
}

DEVI v8f wmma_bf16(v16bf a, v16bf b, v8f c) {
  return __builtin_amdgcn_wmma_f32_16x16x32_bf16(
      false, a, false, b, (short)0, c, false, false);
}

// ---------- fragment loaders ----------

// A fragment (16x32 bf16): lane r=lane&15 holds row M=r; kh=lane>>4 selects
// interleaved K blocks (kh*8..+7 and 16+kh*8..+7) per ISA 16-bit A layout.
DEVI v16bf frag_a_bf16(const __bf16* base, int ld, int row0, int k0, int lane) {
  const int r = lane & 15, kh = lane >> 4;
  const __bf16* p = base + (size_t)(row0 + r) * ld + k0;
  v8bf lo = *(const v8bf*)(p + kh * 8);
  v8bf hi = *(const v8bf*)(p + 16 + kh * 8);
  return __builtin_shufflevector(lo, hi, 0,1,2,3,4,5,6,7,8,9,10,11,12,13,14,15);
}

// A fragment from f32 source: pair-pack conversion into u32 lanes.
DEVI v16bf frag_a_f32(const float* base, int ld, int row0, int k0, int lane) {
  const int r = lane & 15, kh = lane >> 4;
  const float* p = base + (size_t)(row0 + r) * ld + k0 + kh * 8;
  v8u u;
#pragma unroll
  for (int w = 0; w < 4; ++w) {
    u[w]     = pk2bf(p[2 * w],      p[2 * w + 1]);
    u[4 + w] = pk2bf(p[16 + 2 * w], p[16 + 2 * w + 1]);
  }
  return __builtin_bit_cast(v16bf, u);
}

// B fragment (32x16 bf16): lane holds column N=lane&15 (a row of the row-major
// NxK operand), K blocks kh*8..+7 and 16+kh*8..+7 (same order as A).
DEVI v16bf frag_b_bf16(const __bf16* base, int ld, int col0, int k0, int lane) {
  const int r = lane & 15, kh = lane >> 4;
  const __bf16* p = base + (size_t)(col0 + r) * ld + k0;
  v8bf lo = *(const v8bf*)(p + kh * 8);
  v8bf hi = *(const v8bf*)(p + 16 + kh * 8);
  return __builtin_shufflevector(lo, hi, 0,1,2,3,4,5,6,7,8,9,10,11,12,13,14,15);
}

// ---------- 16-lane row reductions via DPP16 (no DS-unit traffic) ----------
// xor masks {1,2,7,15} span lane-id bits 0..3: full 16-lane butterfly.

template <int CTRL>
DEVI float dpp_mov_f32(float x) {
  int xi = __builtin_bit_cast(int, x);
  int r  = __builtin_amdgcn_update_dpp(xi, xi, CTRL, 0xF, 0xF, true);
  return __builtin_bit_cast(float, r);
}

DEVI float red_max16(float v) {
  v = fmaxf(v, dpp_mov_f32<0xB1>(v));    // quad_perm [1,0,3,2]  : xor 1
  v = fmaxf(v, dpp_mov_f32<0x4E>(v));    // quad_perm [2,3,0,1]  : xor 2
  v = fmaxf(v, dpp_mov_f32<0x141>(v));   // row_half_mirror      : xor 7
  v = fmaxf(v, dpp_mov_f32<0x140>(v));   // row_mirror           : xor 15
  return v;
}
DEVI float red_sum16(float v) {
  v += dpp_mov_f32<0xB1>(v);
  v += dpp_mov_f32<0x4E>(v);
  v += dpp_mov_f32<0x141>(v);
  v += dpp_mov_f32<0x140>(v);
  return v;
}

// ---------- weight pre-conversion: f32 -> bf16, 8 elems/thread ----------

__global__ __launch_bounds__(256) void cvt_bf16_kernel(
    const float* __restrict__ in, __bf16* __restrict__ out) {
  const size_t t = (size_t)blockIdx.x * 256 + threadIdx.x;
  const v4f a = ((const v4f*)in)[2 * t];
  const v4f b = ((const v4f*)in)[2 * t + 1];
  v4u o;
  o[0] = pk2bf(a[0], a[1]);
  o[1] = pk2bf(a[2], a[3]);
  o[2] = pk2bf(b[0], b[1]);
  o[3] = pk2bf(b[2], b[3]);
  ((v4u*)out)[t] = o;
}

// ---------- projection GEMM: Y = X @ W.T  (M=8192, N=1024, K=1024) ----------
// Wave computes 32(M) x 64(N); 8 waves/WG -> 256 x 64 macro tile.
// MODE 0: X f32, out bf16 [B,H,S,D], scale 1/sqrt(D)      (Q)
// MODE 1: X f32, out bf16 [B,H,S,D]                        (K)
// MODE 2: X f32, out bf16 [B,H,D,S'] key-interleaved V^T   (V)
// MODE 3: X bf16, out f32 row-major [M,E]                  (final Wo proj)
template <int MODE>
__global__ __launch_bounds__(256) void proj_kernel(
    const void* __restrict__ Xv, const __bf16* __restrict__ Wb,
    void* __restrict__ Out, float scale) {
  const int lane = threadIdx.x & 31;
  const int wv   = threadIdx.x >> 5;
  const int m0   = blockIdx.y * 256 + wv * 32;
  const int n0   = blockIdx.x * 64;

  v8f acc[2][4] = {};

  for (int kk = 0; kk < Ec; kk += 32) {
    v16bf a[2];
#pragma unroll
    for (int mt = 0; mt < 2; ++mt) {
      if (MODE == 3) a[mt] = frag_a_bf16((const __bf16*)Xv, Ec, m0 + mt * 16, kk, lane);
      else           a[mt] = frag_a_f32((const float*)Xv, Ec, m0 + mt * 16, kk, lane);
    }
    if (kk + 32 < Ec) {   // global_prefetch_b8 hint for next A tile
      if (MODE == 3)
        __builtin_prefetch((const __bf16*)Xv + (size_t)(m0 + (lane & 31)) * Ec + kk + 32, 0, 1);
      else
        __builtin_prefetch((const float*)Xv + (size_t)(m0 + (lane & 31)) * Ec + kk + 32, 0, 1);
    }
#pragma unroll
    for (int t = 0; t < 4; ++t) {
      const v16bf bb = frag_b_bf16(Wb, Ec, n0 + t * 16, kk, lane);
#pragma unroll
      for (int mt = 0; mt < 2; ++mt)
        acc[mt][t] = wmma_bf16(a[mt], bb, acc[mt][t]);
    }
  }

  const int nl = lane & 15, chh = lane >> 4;
#pragma unroll
  for (int mt = 0; mt < 2; ++mt) {
#pragma unroll
    for (int t = 0; t < 4; ++t) {
#pragma unroll
      for (int i = 0; i < 8; ++i) {
        const int mrow = m0 + mt * 16 + i + 8 * chh;  // C layout: M = i + 8*(lane/16)
        const int n    = n0 + t * 16 + nl;            // N = lane&15
        const float v  = acc[mt][t][i] * scale;
        if (MODE == 3) {
          ((float*)Out)[(size_t)mrow * Ec + n] = v;
        } else {
          const int bb = mrow >> 12;                  // / S
          const int s  = mrow & (Sc - 1);
          const int hh = n >> 6;                      // / D
          const int d  = n & (Dc - 1);
          size_t idx;
          if (MODE == 2) {
            // key-interleave within each 32-group so the flash kernel can
            // store P as packed dwords: pos = 2*(s%16) + ((s/16)%2)
            const int sp = (s & ~31) | ((s & 15) << 1) | ((s >> 4) & 1);
            idx = (((size_t)bb * Hc + hh) * Dc + d) * Sc + sp;   // [B,H,D,S']
          } else {
            idx = (((size_t)bb * Hc + hh) * Sc + s) * Dc + d;    // [B,H,S,D]
          }
          ((__bf16*)Out)[idx] = f2bf(v);
        }
      }
    }
  }
}

// ---------- causal flash attention ----------
// Grid: (S/64, H, B); 4 waves/WG; wave owns 16 query rows, streams 64-key
// chunks. Only the last chunk straddles the diagonal -> masked tail.

template <bool MASKED>
DEVI void attn_chunk(int kk, int q0, int lane,
                     const __bf16* Kb, const __bf16* Vb,
                     const v16bf& aq0, const v16bf& aq1,
                     float* mrun, float* lrun, v8f* o,
                     unsigned* ldsT /* this wave's 16x32-dword P tile */) {
  const int nl = lane & 15, chh = lane >> 4;

  // scores: S[16q x 64k] = Q(16x64) @ K_chunk^T  (8 WMMAs)
  v8f s[4] = {};
#pragma unroll
  for (int t = 0; t < 4; ++t) {
    s[t] = wmma_bf16(aq0, frag_b_bf16(Kb, Dc, kk + t * 16, 0,  lane), s[t]);
    s[t] = wmma_bf16(aq1, frag_b_bf16(Kb, Dc, kk + t * 16, 32, lane), s[t]);
  }

  // online softmax per C-fragment row
#pragma unroll
  for (int i = 0; i < 8; ++i) {
    float v0 = s[0][i], v1 = s[1][i], v2 = s[2][i], v3 = s[3][i];
    if (MASKED) {
      const int qg = q0 + i + 8 * chh;
      v0 = (kk + nl      <= qg) ? v0 : -3.0e38f;
      v1 = (kk + 16 + nl <= qg) ? v1 : -3.0e38f;
      v2 = (kk + 32 + nl <= qg) ? v2 : -3.0e38f;
      v3 = (kk + 48 + nl <= qg) ? v3 : -3.0e38f;
    }
    const float mx   = red_max16(fmaxf(fmaxf(v0, v1), fmaxf(v2, v3)));
    const float mnew = fmaxf(mrun[i], mx);
    const float sc   = __expf(mrun[i] - mnew);
    const float e0 = __expf(v0 - mnew), e1 = __expf(v1 - mnew);
    const float e2 = __expf(v2 - mnew), e3 = __expf(v3 - mnew);
    lrun[i] = lrun[i] * sc + red_sum16((e0 + e1) + (e2 + e3));
    mrun[i] = mnew;
#pragma unroll
    for (int t = 0; t < 4; ++t) o[t][i] *= sc;
    // packed dword P store; key (16t+nl) of half g lands at tile position
    // 32g + 2*nl + (t&1), matching the key-interleaved V^T layout.
    const int row = i + 8 * chh;
    ldsT[row * 32 + nl]      = pk2bf(e0, e1);   // keys kk+nl,    kk+16+nl
    ldsT[row * 32 + 16 + nl] = pk2bf(e2, e3);   // keys kk+32+nl, kk+48+nl
  }
  asm volatile("" ::: "memory");   // compiler-order LDS store -> load

  // O += P(16x64) @ V_chunk(64x64): reload P in A-layout, 8 WMMAs
  const __bf16* Pt = (const __bf16*)ldsT;
  const v16bf ap0 = frag_a_bf16(Pt, 64, 0, 0,  lane);
  const v16bf ap1 = frag_a_bf16(Pt, 64, 0, 32, lane);
#pragma unroll
  for (int t = 0; t < 4; ++t) {
    o[t] = wmma_bf16(ap0, frag_b_bf16(Vb, Sc, t * 16, kk,      lane), o[t]);
    o[t] = wmma_bf16(ap1, frag_b_bf16(Vb, Sc, t * 16, kk + 32, lane), o[t]);
  }
  asm volatile("" ::: "memory");
}

__global__ __launch_bounds__(128) void flash_attn_kernel(
    const __bf16* __restrict__ Qp, const __bf16* __restrict__ Kp,
    const __bf16* __restrict__ Vt, __bf16* __restrict__ Ob) {
  __shared__ unsigned ldsP[4][16][32];   // per-wave P tile (packed bf16 pairs)

  const int lane = threadIdx.x & 31;
  const int wv   = threadIdx.x >> 5;
  const int b    = blockIdx.z;
  const int h    = blockIdx.y;
  const int q0   = blockIdx.x * 64 + wv * 16;

  const __bf16* Qb = Qp + (size_t)(b * Hc + h) * Sc * Dc;   // [S,D]
  const __bf16* Kb = Kp + (size_t)(b * Hc + h) * Sc * Dc;   // [S,D]
  const __bf16* Vb = Vt + (size_t)(b * Hc + h) * Dc * Sc;   // [D,S'] interleaved

  const v16bf aq0 = frag_a_bf16(Qb, Dc, q0, 0,  lane);
  const v16bf aq1 = frag_a_bf16(Qb, Dc, q0, 32, lane);

  float mrun[8], lrun[8];
  v8f o[4] = {};
#pragma unroll
  for (int i = 0; i < 8; ++i) { mrun[i] = -3.0e38f; lrun[i] = 0.0f; }

  unsigned* ldsT = &ldsP[wv][0][0];
  const int nch = (q0 + 16 + 63) >> 6;   // 64-key chunks covering keys <= q0+15

  for (int c = 0; c < nch - 1; ++c)      // fully below diagonal: no masking
    attn_chunk<false>(c * 64, q0, lane, Kb, Vb, aq0, aq1, mrun, lrun, o, ldsT);
  attn_chunk<true>((nch - 1) * 64, q0, lane, Kb, Vb, aq0, aq1, mrun, lrun, o, ldsT);

  // normalize; store bf16 attention output as [B*S, E] for the Wo GEMM
  const int nl = lane & 15, chh = lane >> 4;
#pragma unroll
  for (int t = 0; t < 4; ++t) {
#pragma unroll
    for (int i = 0; i < 8; ++i) {
      const int srow = q0 + i + 8 * chh;
      const float val = o[t][i] / lrun[i];
      Ob[(size_t)(b * Sc + srow) * Ec + h * Dc + t * 16 + nl] = f2bf(val);
    }
  }
}

// ---------- launch ----------

extern "C" void kernel_launch(void* const* d_in, const int* in_sizes, int n_in,
                              void* d_out, int out_size, void* d_ws, size_t ws_size,
                              hipStream_t stream) {
  (void)in_sizes; (void)n_in; (void)out_size; (void)ws_size;
  // setup_inputs order: k, v, q, Wk, Wv, Wq, Wo
  const float* k  = (const float*)d_in[0];
  const float* v  = (const float*)d_in[1];
  const float* q  = (const float*)d_in[2];
  const float* Wk = (const float*)d_in[3];
  const float* Wv = (const float*)d_in[4];
  const float* Wq = (const float*)d_in[5];
  const float* Wo = (const float*)d_in[6];

  const size_t nBHSD = (size_t)Bc * Hc * Sc * Dc;   // 8M elems
  const size_t nEE   = (size_t)Ec * Ec;             // 1M elems
  __bf16* Qp  = (__bf16*)d_ws;         // [B,H,S,D]   16 MB
  __bf16* Kp  = Qp + nBHSD;            // [B,H,S,D]   16 MB
  __bf16* Vt  = Kp + nBHSD;            // [B,H,D,S']  16 MB
  __bf16* Ob  = Vt + nBHSD;            // [B*S, E]    16 MB
  __bf16* Wqb = Ob + nBHSD;            // bf16 weights, 2 MB each (total 72 MB)
  __bf16* Wkb = Wqb + nEE;
  __bf16* Wvb = Wkb + nEE;
  __bf16* Wob = Wvb + nEE;

  const int cvtBlocks = (int)(nEE / (256 * 8));     // 512
  cvt_bf16_kernel<<<cvtBlocks, 256, 0, stream>>>(Wq, Wqb);
  cvt_bf16_kernel<<<cvtBlocks, 256, 0, stream>>>(Wk, Wkb);
  cvt_bf16_kernel<<<cvtBlocks, 256, 0, stream>>>(Wv, Wvb);
  cvt_bf16_kernel<<<cvtBlocks, 256, 0, stream>>>(Wo, Wob);

  const dim3 gp(Ec / 64, Mrows / 256);              // (16, 32), 256 threads
  proj_kernel<0><<<gp, 256, 0, stream>>>(q, Wqb, Qp, 0.125f);  // 1/sqrt(64)
  proj_kernel<1><<<gp, 256, 0, stream>>>(k, Wkb, Kp, 1.0f);
  proj_kernel<2><<<gp, 256, 0, stream>>>(v, Wvb, Vt, 1.0f);

  flash_attn_kernel<<<dim3(Sc / 64, Hc, Bc), 128, 0, stream>>>(Qp, Kp, Vt, Ob);

  proj_kernel<3><<<gp, 256, 0, stream>>>(Ob, Wob, d_out, 1.0f);
}
